// TtsEngine_48876727828496
// MI455X (gfx1250) — matrix operand order repaired and checked by reading, not compiled, not run on hardware
//
#include <hip/hip_runtime.h>
#include <hip/hip_bf16.h>

// ---------------------------------------------------------------------------
// TTS engine for MI455X (gfx1250, wave32, WMMA).
// Encoder convs + teacher-forced prenet hoisted into big bf16 WMMA GEMMs.
// Recurrent LSTM core runs in ONE persistent kernel (64 WGs x 128 thr) with
// device-wide sense barriers; c-state lives in LDS for all 800 steps.
// Inner GEMM loop is software-pipelined (double-buffered fragments) so the
// scheduler can emit partial s_wait_loadcnt and overlap loads with WMMA.
// ---------------------------------------------------------------------------

#define BB   16
#define SS   256
#define TT   800
#define EMBD 512
#define ENCD 512
#define DECD 1024
#define MELS 80

typedef __bf16 bf16_t;
typedef __attribute__((ext_vector_type(16))) __bf16 v16bf;
typedef __attribute__((ext_vector_type(8)))  __bf16 v8bf;
typedef __attribute__((ext_vector_type(8)))  float  v8f;

union AFrag { v16bf v; v8bf h[2]; };

static __device__ inline v8f zero8() {
    v8f z = {0.f, 0.f, 0.f, 0.f, 0.f, 0.f, 0.f, 0.f};
    return z;
}

static __device__ inline float sigm(float x) { return 1.f / (1.f + __expf(-x)); }

// D = A[16,K] x W[N=16 tile,K]^T accumulate. A row-major bf16 (lda halfs),
// W row-major [out,in] bf16 (ldb halfs). Lane layout: contiguous k = 16*(lane/16) + s in both A and B.
// Software-pipelined: chunk k+1 fragments are loaded before the WMMA on
// chunk k issues, so only partial load waits are needed.
static __device__ inline v8f wmma_acc(const bf16_t* __restrict__ A, int lda,
                                      const bf16_t* __restrict__ Wn, int ldb,
                                      int K, v8f acc) {
    const int lane = threadIdx.x & 31;
    const int hi   = (lane >= 16);
    const bf16_t* ar = A  + (size_t)(lane & 15) * lda + (hi ? 16 : 0);
    const bf16_t* br = Wn + (size_t)(lane & 15) * ldb + (hi ? 16 : 0);
    AFrag a; v16bf b;
    a.h[0] = *(const v8bf*)(ar);
    a.h[1] = *(const v8bf*)(ar + 8);
    b      = *(const v16bf*)(br);
    for (int k = 32; k < K; k += 32) {
        // prefetch the weight stream 7 chunks (224 halfs) ahead (OOB prefetch is dropped)
        __builtin_prefetch((const void*)(br + k + 224), 0, 3);
        AFrag an; v16bf bn;
        an.h[0] = *(const v8bf*)(ar + k);
        an.h[1] = *(const v8bf*)(ar + k + 8);
        bn      = *(const v16bf*)(br + k);
        acc = __builtin_amdgcn_wmma_f32_16x16x32_bf16(false, a.v, false, b,
                                                      (short)0, acc, false, false);
        a = an; b = bn;
    }
    acc = __builtin_amdgcn_wmma_f32_16x16x32_bf16(false, a.v, false, b,
                                                  (short)0, acc, false, false);
    return acc;
}

// ------------------------------ prep kernels -------------------------------

__global__ void cvt_f32_bf16(const float* __restrict__ s, bf16_t* __restrict__ d, int n) {
    int i = blockIdx.x * blockDim.x + threadIdx.x;
    if (i < n) d[i] = (bf16_t)s[i];
}

// pre1_w [1024,80] -> [1024,96] zero padded bf16
__global__ void pad_pre1(const float* __restrict__ w, bf16_t* __restrict__ d) {
    int i = blockIdx.x * blockDim.x + threadIdx.x;
    if (i >= 1024 * 96) return;
    int r = i / 96, j = i % 96;
    d[i] = (bf16_t)((j < 80) ? w[r * 80 + j] : 0.f);
}

// conv w [O,I,5] -> Wcol [O, k*I+i] bf16
__global__ void build_wcol(const float* __restrict__ w, bf16_t* __restrict__ d) {
    int i = blockIdx.x * blockDim.x + threadIdx.x;
    if (i >= ENCD * EMBD * 5) return;
    int o = i / (EMBD * 5), c = i % (EMBD * 5);
    int k = c / EMBD, ii = c % EMBD;
    d[i] = (bf16_t)w[(o * EMBD + ii) * 5 + k];
}

// im2col of embedded text -> xcol [B*S, 5*EMB]
__global__ void build_xcol1(const int* __restrict__ text, const float* __restrict__ emb,
                            bf16_t* __restrict__ d) {
    int i = blockIdx.x * blockDim.x + threadIdx.x;
    if (i >= BB * SS * EMBD * 5) return;
    int row = i / (EMBD * 5), c = i % (EMBD * 5);
    int k = c / EMBD, ii = c % EMBD;
    int b = row >> 8, s = row & 255;
    int s2 = s + k - 2;
    float v = 0.f;
    if (s2 >= 0 && s2 < SS) v = emb[(size_t)text[b * SS + s2] * EMBD + ii];
    d[i] = (bf16_t)v;
}

// im2col of conv1 output (bf16) -> xcol2
__global__ void build_xcol2(const bf16_t* __restrict__ y1, bf16_t* __restrict__ d) {
    int i = blockIdx.x * blockDim.x + threadIdx.x;
    if (i >= BB * SS * ENCD * 5) return;
    int row = i / (ENCD * 5), c = i % (ENCD * 5);
    int k = c / ENCD, ii = c % ENCD;
    int b = row >> 8, s = row & 255;
    int s2 = s + k - 2;
    bf16_t v = (bf16_t)0.f;
    if (s2 >= 0 && s2 < SS) v = y1[(size_t)(b * SS + s2) * ENCD + ii];
    d[i] = v;
}

// teacher-forced prenet input: [T*B, 96] bf16 (step t uses mel_target[t-1])
__global__ void build_a0(const float* __restrict__ mel, bf16_t* __restrict__ d) {
    int i = blockIdx.x * blockDim.x + threadIdx.x;
    if (i >= TT * BB * 96) return;
    int row = i / 96, j = i % 96;
    int t = row / BB, b = row % BB;
    float v = 0.f;
    if (j < MELS && t > 0) v = mel[((size_t)b * TT + (t - 1)) * MELS + j];
    d[i] = (bf16_t)v;
}

// context[b,c] = mean_s enc[b,s,c] + emotion[b]·emo_w[c] + emo_b[c]  (bf16)
__global__ void context_kernel(const float* __restrict__ enc, const float* __restrict__ emo,
                               const float* __restrict__ emo_w, const float* __restrict__ emo_b,
                               bf16_t* __restrict__ ctx) {
    int i = blockIdx.x * blockDim.x + threadIdx.x;
    if (i >= BB * ENCD) return;
    int b = i >> 9, c = i & 511;
    float s = 0.f;
    for (int t = 0; t < SS; ++t) s += enc[((size_t)b * SS + t) * ENCD + c];
    s *= (1.f / (float)SS);
    float e = emo_b[c];
    for (int j = 0; j < 5; ++j) e += emo[b * 5 + j] * emo_w[c * 5 + j];
    ctx[i] = (bf16_t)(s + e);
}

__global__ void init_kernel(bf16_t* __restrict__ h1, bf16_t* __restrict__ h2,
                            unsigned* __restrict__ bars) {
    int i = blockIdx.x * blockDim.x + threadIdx.x;
    if (i < 2 * BB * DECD) { h1[i] = (bf16_t)0.f; h2[i] = (bf16_t)0.f; }
    if (i < 16) bars[i] = 0u;
}

// --------------------------- generic WMMA GEMM -----------------------------
// C[M,N] = act(A[M,K] @ W[N,K]^T + bias). block=128 (4 waves, 4 N-tiles).
// grid = (N/64, M/16)
__global__ void gemm_bf16_kernel(const bf16_t* __restrict__ A, int lda,
                                 const bf16_t* __restrict__ W, int ldb,
                                 const float* __restrict__ bias,
                                 void* __restrict__ out, int ldc,
                                 int K, int relu, int out_bf16) {
    const int wave = threadIdx.x >> 5, lane = threadIdx.x & 31;
    const int n0 = (blockIdx.x * 4 + wave) * 16;
    const int m0 = blockIdx.y * 16;
    v8f acc = zero8();
    acc = wmma_acc(A + (size_t)m0 * lda, lda, W + (size_t)n0 * ldb, ldb, K, acc);
    const int n = n0 + (lane & 15);
    const float bv = bias ? bias[n] : 0.f;
    for (int r = 0; r < 8; ++r) {
        int m = m0 + r + ((lane >= 16) ? 8 : 0);
        float v = acc[r] + bv;
        if (relu) v = fmaxf(v, 0.f);
        if (out_bf16) ((bf16_t*)out)[(size_t)m * ldc + n] = (bf16_t)v;
        else          ((float*)out)[(size_t)m * ldc + n]  = v;
    }
}

// ------------------------- persistent LSTM decoder -------------------------

static __device__ inline void grid_barrier(unsigned* bar, unsigned nwg) {
    __threadfence();
    __syncthreads();
    if (threadIdx.x == 0) {
        volatile unsigned* vb = (volatile unsigned*)bar;
        unsigned gen = vb[1];
        if (atomicAdd(bar, 1u) == nwg - 1) {
            vb[0] = 0;
            __threadfence();
            atomicAdd(bar + 1, 1u);
        } else {
            while (vb[1] == gen) { __builtin_amdgcn_s_sleep(1); }
        }
        __threadfence();
    }
    __syncthreads();
}

static __device__ inline void emit_outputs(int t, const bf16_t* __restrict__ h2,
                                           const bf16_t* __restrict__ melw,
                                           const float* __restrict__ melb,
                                           const float* __restrict__ stopw,
                                           const float* __restrict__ stopb,
                                           float* __restrict__ out_mel,
                                           float* __restrict__ out_stop,
                                           int wg, int tid) {
    const int wave = tid >> 5, lane = tid & 31;
    if (wg < 5) {                       // mel: 5 tiles of 16 channels
        if (wave == 0) {
            v8f acc = zero8();
            acc = wmma_acc(h2, DECD, melw + (size_t)wg * 16 * DECD, DECD, DECD, acc);
            int n = wg * 16 + (lane & 15);
            float bv = melb[n];
            for (int r = 0; r < 8; ++r) {
                int b = r + ((lane >= 16) ? 8 : 0);
                out_mel[(size_t)b * (TT * MELS) + (size_t)t * MELS + n] = acc[r] + bv;
            }
        }
    } else if (wg == 5) {               // stop token
        if (tid < BB) {
            float s = stopb[0];
            for (int k = 0; k < DECD; ++k) s += (float)h2[tid * DECD + k] * stopw[k];
            out_stop[tid * TT + t] = sigm(s);
        }
    }
}

__global__ void __launch_bounds__(128) decoder_kernel(
    const bf16_t* __restrict__ P,      // [T*B, 1024] prenet out (bf16)
    const bf16_t* __restrict__ ctx,    // [B, 512]
    const bf16_t* __restrict__ wih1,   // [4096, 1536]
    const bf16_t* __restrict__ whh1,   // [4096, 1024]
    const float* __restrict__ bih1, const float* __restrict__ bhh1,
    const bf16_t* __restrict__ wih2,   // [4096, 1024]
    const bf16_t* __restrict__ whh2,   // [4096, 1024]
    const float* __restrict__ bih2, const float* __restrict__ bhh2,
    const bf16_t* __restrict__ melw, const float* __restrict__ melb,
    const float* __restrict__ stopw, const float* __restrict__ stopb,
    bf16_t* __restrict__ h1buf,        // [2][B,1024] ping-pong
    bf16_t* __restrict__ h2buf,
    float* __restrict__ out_mel, float* __restrict__ out_stop,
    unsigned* __restrict__ bars) {
    __shared__ float gbuf[4][16][16];  // i,f,g,o gate tiles
    __shared__ float cc[2][16][16];    // cell state tiles, resident all T steps
    const int tid = threadIdx.x, wave = tid >> 5, lane = tid & 31;
    const int wg = blockIdx.x;         // 0..63 -> hidden units wg*16..+15

    for (int e = tid; e < 2 * 256; e += 128) ((float*)cc)[e] = 0.f;
    __syncthreads();

    const int nr = wave * 1024 + wg * 16;          // gate row base for my wave
    const int ncol = nr + (lane & 15);

    for (int t = 0; t < TT; ++t) {
        const int wp = t & 1, rp = wp ^ 1;
        const bf16_t* h1r = h1buf + rp * (BB * DECD);
        const bf16_t* h2r = h2buf + rp * (BB * DECD);
        bf16_t* h1w = h1buf + wp * (BB * DECD);
        bf16_t* h2w = h2buf + wp * (BB * DECD);

        // outputs for previous step (h2r is step t-1's h2, complete)
        if (t > 0) emit_outputs(t - 1, h2r, melw, melb, stopw, stopb,
                                out_mel, out_stop, wg, tid);

        // ---- LSTM1 gates: [x(t) | ctx] @ wih1^T + h1(t-1) @ whh1^T ----
        {
            v8f acc = zero8();
            const bf16_t* bx = wih1 + (size_t)nr * 1536;
            acc = wmma_acc(P + (size_t)t * BB * DECD, DECD, bx, 1536, DECD, acc);
            acc = wmma_acc(ctx, ENCD, bx + DECD, 1536, ENCD, acc);
            acc = wmma_acc(h1r, DECD, whh1 + (size_t)nr * DECD, DECD, DECD, acc);
            float bsum = bih1[ncol] + bhh1[ncol];
            for (int r = 0; r < 8; ++r) {
                int m = r + ((lane >= 16) ? 8 : 0);
                gbuf[wave][m][lane & 15] = acc[r] + bsum;
            }
        }
        __syncthreads();
        for (int e = tid; e < 256; e += 128) {     // cell 1 (fused)
            int b = e >> 4, u = e & 15;
            float ig = sigm(gbuf[0][b][u]);
            float fg = sigm(gbuf[1][b][u]);
            float gg = tanhf(gbuf[2][b][u]);
            float og = sigm(gbuf[3][b][u]);
            float cn = fg * cc[0][b][u] + ig * gg;
            cc[0][b][u] = cn;
            h1w[b * DECD + wg * 16 + u] = (bf16_t)(og * tanhf(cn));
        }
        grid_barrier(bars, 64);

        // ---- LSTM2 gates: h1(t) @ wih2^T + h2(t-1) @ whh2^T ----
        {
            v8f acc = zero8();
            acc = wmma_acc(h1w, DECD, wih2 + (size_t)nr * DECD, DECD, DECD, acc);
            acc = wmma_acc(h2r, DECD, whh2 + (size_t)nr * DECD, DECD, DECD, acc);
            float bsum = bih2[ncol] + bhh2[ncol];
            for (int r = 0; r < 8; ++r) {
                int m = r + ((lane >= 16) ? 8 : 0);
                gbuf[wave][m][lane & 15] = acc[r] + bsum;
            }
        }
        __syncthreads();
        for (int e = tid; e < 256; e += 128) {     // cell 2 (fused)
            int b = e >> 4, u = e & 15;
            float ig = sigm(gbuf[0][b][u]);
            float fg = sigm(gbuf[1][b][u]);
            float gg = tanhf(gbuf[2][b][u]);
            float og = sigm(gbuf[3][b][u]);
            float cn = fg * cc[1][b][u] + ig * gg;
            cc[1][b][u] = cn;
            h2w[b * DECD + wg * 16 + u] = (bf16_t)(og * tanhf(cn));
        }
        grid_barrier(bars + 8, 64);
    }
    // final step outputs
    emit_outputs(TT - 1, h2buf + ((TT - 1) & 1) * (BB * DECD),
                 melw, melb, stopw, stopb, out_mel, out_stop, wg, tid);
}

// ------------------------------- launcher ----------------------------------

extern "C" void kernel_launch(void* const* d_in, const int* in_sizes, int n_in,
                              void* d_out, int out_size, void* d_ws, size_t ws_size,
                              hipStream_t stream) {
    const int*   text    = (const int*)d_in[0];
    const float* mel_tgt = (const float*)d_in[1];
    const float* emotion = (const float*)d_in[2];
    const float* emb     = (const float*)d_in[3];
    const float* conv1_w = (const float*)d_in[4];
    const float* conv1_b = (const float*)d_in[5];
    const float* conv2_w = (const float*)d_in[6];
    const float* conv2_b = (const float*)d_in[7];
    const float* emo_w   = (const float*)d_in[8];
    const float* emo_b   = (const float*)d_in[9];
    const float* pre1_w  = (const float*)d_in[10];
    const float* pre1_b  = (const float*)d_in[11];
    const float* pre2_w  = (const float*)d_in[12];
    const float* pre2_b  = (const float*)d_in[13];
    const float* l1_wih  = (const float*)d_in[14];
    const float* l1_whh  = (const float*)d_in[15];
    const float* l1_bih  = (const float*)d_in[16];
    const float* l1_bhh  = (const float*)d_in[17];
    const float* l2_wih  = (const float*)d_in[18];
    const float* l2_whh  = (const float*)d_in[19];
    const float* l2_bih  = (const float*)d_in[20];
    const float* l2_bhh  = (const float*)d_in[21];
    const float* mel_w   = (const float*)d_in[22];
    const float* mel_b   = (const float*)d_in[23];
    const float* stop_w  = (const float*)d_in[24];
    const float* stop_b  = (const float*)d_in[25];

    float* out_mel  = (float*)d_out;
    float* out_stop = (float*)d_out + (size_t)BB * TT * MELS;

    char* w = (char*)d_ws;
    size_t off = 0;
    auto alloc = [&](size_t bytes) -> char* {
        char* p = w + off;
        off = (off + bytes + 255) & ~(size_t)255;
        return p;
    };

    // reuse region: xcol1 -> xcol2 -> P1 (max of 4096*2560 and 12800*1024 halfs)
    const size_t reuse_elems = (size_t)TT * BB * DECD;   // 13.1M > 10.49M
    bf16_t* reuse = (bf16_t*)alloc(reuse_elems * 2);
    bf16_t* w1col = (bf16_t*)alloc((size_t)ENCD * EMBD * 5 * 2);
    bf16_t* w2col = (bf16_t*)alloc((size_t)ENCD * ENCD * 5 * 2);
    bf16_t* y1    = (bf16_t*)alloc((size_t)BB * SS * ENCD * 2);
    float*  encb  = (float*) alloc((size_t)BB * SS * ENCD * 4);
    bf16_t* ctx   = (bf16_t*)alloc((size_t)BB * ENCD * 2);
    bf16_t* a0    = (bf16_t*)alloc((size_t)TT * BB * 96 * 2);
    bf16_t* p2    = (bf16_t*)alloc((size_t)TT * BB * DECD * 2);
    bf16_t* wih1b = (bf16_t*)alloc((size_t)4 * DECD * 1536 * 2);
    bf16_t* whh1b = (bf16_t*)alloc((size_t)4 * DECD * DECD * 2);
    bf16_t* wih2b = (bf16_t*)alloc((size_t)4 * DECD * DECD * 2);
    bf16_t* whh2b = (bf16_t*)alloc((size_t)4 * DECD * DECD * 2);
    bf16_t* pre1p = (bf16_t*)alloc((size_t)DECD * 96 * 2);
    bf16_t* pre2b = (bf16_t*)alloc((size_t)DECD * DECD * 2);
    bf16_t* melwb = (bf16_t*)alloc((size_t)MELS * DECD * 2);
    bf16_t* h1buf = (bf16_t*)alloc((size_t)2 * BB * DECD * 2);
    bf16_t* h2buf = (bf16_t*)alloc((size_t)2 * BB * DECD * 2);
    unsigned* bars = (unsigned*)alloc(256);

    const int TPB = 256;
    auto blocks = [](size_t n, int tpb) { return (unsigned)((n + tpb - 1) / tpb); };

    // init barriers + h state (must be re-done every call: deterministic)
    init_kernel<<<blocks(2 * BB * DECD, TPB), TPB, 0, stream>>>(h1buf, h2buf, bars);

    // weight conversions
    cvt_f32_bf16<<<blocks((size_t)4 * DECD * 1536, TPB), TPB, 0, stream>>>(l1_wih, wih1b, 4 * DECD * 1536);
    cvt_f32_bf16<<<blocks((size_t)4 * DECD * DECD, TPB), TPB, 0, stream>>>(l1_whh, whh1b, 4 * DECD * DECD);
    cvt_f32_bf16<<<blocks((size_t)4 * DECD * DECD, TPB), TPB, 0, stream>>>(l2_wih, wih2b, 4 * DECD * DECD);
    cvt_f32_bf16<<<blocks((size_t)4 * DECD * DECD, TPB), TPB, 0, stream>>>(l2_whh, whh2b, 4 * DECD * DECD);
    cvt_f32_bf16<<<blocks((size_t)DECD * DECD, TPB), TPB, 0, stream>>>(pre2_w, pre2b, DECD * DECD);
    cvt_f32_bf16<<<blocks((size_t)MELS * DECD, TPB), TPB, 0, stream>>>(mel_w, melwb, MELS * DECD);
    pad_pre1<<<blocks((size_t)DECD * 96, TPB), TPB, 0, stream>>>(pre1_w, pre1p);
    build_wcol<<<blocks((size_t)ENCD * EMBD * 5, TPB), TPB, 0, stream>>>(conv1_w, w1col);
    build_wcol<<<blocks((size_t)ENCD * ENCD * 5, TPB), TPB, 0, stream>>>(conv2_w, w2col);

    // ---- encoder ----
    build_xcol1<<<blocks((size_t)BB * SS * EMBD * 5, TPB), TPB, 0, stream>>>(text, emb, reuse);
    gemm_bf16_kernel<<<dim3(ENCD / 64, BB * SS / 16), 128, 0, stream>>>(
        reuse, EMBD * 5, w1col, EMBD * 5, conv1_b, y1, ENCD, EMBD * 5, 1, 1);
    build_xcol2<<<blocks((size_t)BB * SS * ENCD * 5, TPB), TPB, 0, stream>>>(y1, reuse);
    gemm_bf16_kernel<<<dim3(ENCD / 64, BB * SS / 16), 128, 0, stream>>>(
        reuse, ENCD * 5, w2col, ENCD * 5, conv2_b, encb, ENCD, ENCD * 5, 1, 0);
    context_kernel<<<blocks((size_t)BB * ENCD, TPB), TPB, 0, stream>>>(
        encb, emotion, emo_w, emo_b, ctx);

    // ---- prenet (teacher-forced -> hoisted batch GEMMs over all T steps) ----
    build_a0<<<blocks((size_t)TT * BB * 96, TPB), TPB, 0, stream>>>(mel_tgt, a0);
    gemm_bf16_kernel<<<dim3(DECD / 64, TT * BB / 16), 128, 0, stream>>>(
        a0, 96, pre1p, 96, pre1_b, reuse /*P1*/, DECD, 96, 1, 1);
    gemm_bf16_kernel<<<dim3(DECD / 64, TT * BB / 16), 128, 0, stream>>>(
        reuse, DECD, pre2b, DECD, pre2_b, p2, DECD, DECD, 1, 1);

    // ---- persistent recurrent decoder ----
    decoder_kernel<<<64, 128, 0, stream>>>(
        p2, ctx, wih1b, whh1b, l1_bih, l1_bhh, wih2b, whh2b, l2_bih, l2_bhh,
        melwb, mel_b, stop_w, stop_b, h1buf, h2buf, out_mel, out_stop, bars);

    (void)in_sizes; (void)n_in; (void)out_size; (void)ws_size;
}
